// KANMoELayer_74758200754439
// MI455X (gfx1250) — matrix operand order repaired and checked by reading, not compiled, not run on hardware
//
#include <hip/hip_runtime.h>

typedef _Float16 v8h  __attribute__((ext_vector_type(8)));
typedef _Float16 v16h __attribute__((ext_vector_type(16)));
typedef float    v8f  __attribute__((ext_vector_type(8)));
typedef float    v4f  __attribute__((ext_vector_type(4)));

#define GRIDSZ   10
#define NBASIS   13          // GRIDSZ + degree(3)
#define NIN      256
#define NOUT     256
#define NEXP     8
#define KTOT     (NIN*NEXP)  // 2048
#define MTOT     8192        // B*S
#define MTILE    64          // rows per block (4 WMMA M-tiles)
#define ICHUNK   32          // i per K-chunk -> K-chunk = 256
#define KCHUNK   (ICHUNK*NEXP)
#define NCHUNKS  (NIN/ICHUNK)          // 8
#define ROWPAD   (KCHUNK + 8)          // 264 halves -> 528B stride = +4 banks/row

// ---------------- Kernel 1: softmax gates -> K-major f16 B matrix -------------
// Gp[o][k] with k = i*8 + e, contiguous in k. 256*2048 f16 = 1 MB in d_ws.
__global__ __launch_bounds__(256)
void kan_prep_gates(const float* __restrict__ gw, _Float16* __restrict__ Gp) {
  const int tid = blockIdx.x * 256 + threadIdx.x;      // 65536 total
  const int i = tid & (NIN - 1);
  const int o = tid >> 8;
  const float* p = gw + ((size_t)i * NOUT + o) * NEXP; // gating (I,O,E), E contiguous
  v4f a = *(const v4f*)p;
  v4f b = *(const v4f*)(p + 4);
  float mx = a[0];
  #pragma unroll
  for (int j = 1; j < 4; ++j) mx = fmaxf(mx, a[j]);
  #pragma unroll
  for (int j = 0; j < 4; ++j) mx = fmaxf(mx, b[j]);
  float e[8], s = 0.f;
  #pragma unroll
  for (int j = 0; j < 4; ++j) { e[j]     = __expf(a[j] - mx); s += e[j]; }
  #pragma unroll
  for (int j = 0; j < 4; ++j) { e[j + 4] = __expf(b[j] - mx); s += e[j + 4]; }
  const float inv = 1.0f / s;
  v8h r;
  #pragma unroll
  for (int j = 0; j < 8; ++j) r[j] = (_Float16)(e[j] * inv);
  *(v8h*)(Gp + (size_t)o * KTOT + i * NEXP) = r;       // 16B aligned
}

// ---------------- Kernel 2: fused spline-expert eval + WMMA GEMM -------------
// Block: 256 thr, M=64. Each wave: 4 M-tiles x 2 N-tiles (B frag reused 4x,
// A frags preloaded so 8 WMMAs issue back-to-back behind a single wait).
__global__ __launch_bounds__(256)
void kan_moe_fused(const float* __restrict__ x, const float* __restrict__ coeff,
                   const _Float16* __restrict__ Gp, float* __restrict__ out) {
  __shared__ __align__(16) float    CT[NBASIS][NEXP];   // coeff transposed (13x8)
  __shared__ __align__(16) _Float16 A[MTILE][ROWPAD];   // 64 x 256 f16 panel (+pad)

  const int t    = threadIdx.x;
  const int lane = t & 31;
  const int wave = t >> 5;
  const int m0   = blockIdx.x * MTILE;

  if (t < NBASIS * NEXP) {                              // transpose coeff (E,13)->(13,E)
    const int n = t >> 3, e = t & 7;
    CT[n][e] = coeff[e * NBASIS + n];
  }

  v8f acc[4][2] = {};                                   // [M-tile][N-tile]
  const int nt0 = wave * 2, nt1 = nt0 + 1;
  const int colHalf = lane >> 4;                        // 0: k 0-7,16-23; 1: k 8-15,24-31 (per K-step)
  const int col = lane & 15;

  const int am = t >> 2;                                // phase-A row (0..63)
  const int al = t & 3;                                 // 4 threads per row

  for (int kb = 0; kb < NCHUNKS; ++kb) {
    __syncthreads();   // kb==0: CT ready; else: previous chunk's WMMA readers done
    // ---- Phase A: 8 (m,i) pairs per thread, 8 expert f16 values each ----
    const float* xr = x + (size_t)(m0 + am) * NIN + kb * ICHUNK;
    #pragma unroll
    for (int j = 0; j < 8; ++j) {
      const int ii = al + 4 * j;                        // 0..31 within chunk
      float xv = xr[ii];
      xv = fminf(fmaxf(xv, -1.0f), 1.0f - 1e-6f);
      const float u = (xv + 1.0f) * (GRIDSZ * 0.5f);    // cell coordinate
      int c = (int)u; c = c > GRIDSZ - 1 ? GRIDSZ - 1 : c;
      const float tt = u - (float)c, s1 = 1.0f - tt;
      const float t2 = tt * tt, t3 = t2 * tt;
      const float i6 = 1.0f / 6.0f;
      const float w0 = s1 * s1 * s1 * i6;
      const float w1 = (3.0f * t3 - 6.0f * t2 + 4.0f) * i6;
      const float w2 = (-3.0f * t3 + 3.0f * t2 + 3.0f * tt + 1.0f) * i6;
      const float w3 = t3 * i6;
      const v4f* r0 = (const v4f*)&CT[c][0];
      const v4f* r1 = (const v4f*)&CT[c + 1][0];
      const v4f* r2 = (const v4f*)&CT[c + 2][0];
      const v4f* r3 = (const v4f*)&CT[c + 3][0];
      v4f lo = w0 * r0[0] + w1 * r1[0] + w2 * r2[0] + w3 * r3[0];
      v4f hi = w0 * r0[1] + w1 * r1[1] + w2 * r2[1] + w3 * r3[1];
      v8h rr;
      #pragma unroll
      for (int q = 0; q < 4; ++q) { rr[q] = (_Float16)lo[q]; rr[q + 4] = (_Float16)hi[q]; }
      *(v8h*)&A[am][ii * NEXP] = rr;                    // ds_store_b128
    }
    __syncthreads();
    // ---- Phase B: 8 WMMA K-steps over this 256-wide chunk ----
    #pragma unroll 2
    for (int kk = 0; kk < KCHUNK / 32; ++kk) {
      const int kbase = kk * 32;
      const int kg = kb * KCHUNK + kbase + colHalf * 8;
      const _Float16* b0p = Gp + (size_t)(nt0 * 16 + col) * KTOT + kg;
      const _Float16* b1p = Gp + (size_t)(nt1 * 16 + col) * KTOT + kg;
      if (kk == 0) {                                    // prefetch next chunk's B
        __builtin_prefetch(b0p + KCHUNK, 0, 3);         // global_prefetch_b8
        __builtin_prefetch(b1p + KCHUNK, 0, 3);
      }
      const v8h b0l = *(const v8h*)b0p, b0h = *(const v8h*)(b0p + 16);
      const v8h b1l = *(const v8h*)b1p, b1h = *(const v8h*)(b1p + 16);
      v16h bf0, bf1;
      #pragma unroll
      for (int q = 0; q < 8; ++q) {
        bf0[q] = b0l[q]; bf0[q + 8] = b0h[q];
        bf1[q] = b1l[q]; bf1[q + 8] = b1h[q];
      }
      // Preload all 4 A fragments (distinct regs) -> one clause + single wait
      v16h afr[4];
      #pragma unroll
      for (int mt = 0; mt < 4; ++mt) {
        const _Float16* ap = &A[mt * 16 + col][kbase + colHalf * 8];
        const v8h alo = *(const v8h*)ap;
        const v8h ahi = *(const v8h*)(ap + 16);
        #pragma unroll
        for (int q = 0; q < 8; ++q) { afr[mt][q] = alo[q]; afr[mt][q + 8] = ahi[q]; }
      }
      // 8 back-to-back WMMAs; all accumulators distinct -> no D->A/B hazards
      #pragma unroll
      for (int mt = 0; mt < 4; ++mt) {
        acc[mt][0] = __builtin_amdgcn_wmma_f32_16x16x32_f16(false, afr[mt], false, bf0,
                                                            (short)0, acc[mt][0], false, false);
        acc[mt][1] = __builtin_amdgcn_wmma_f32_16x16x32_f16(false, afr[mt], false, bf1,
                                                            (short)0, acc[mt][1], false, false);
      }
    }
  }
  // ---- Store: (VGPR r, lane) -> M = mt*16 + r + 8*colHalf, N = col ----
  float* op = out + (size_t)m0 * NOUT;
  #pragma unroll
  for (int mt = 0; mt < 4; ++mt) {
    #pragma unroll
    for (int r = 0; r < 8; ++r) {
      const int mm = mt * 16 + r + 8 * colHalf;
      op[(size_t)mm * NOUT + nt0 * 16 + col] = acc[mt][0][r];
      op[(size_t)mm * NOUT + nt1 * 16 + col] = acc[mt][1][r];
    }
  }
}

extern "C" void kernel_launch(void* const* d_in, const int* in_sizes, int n_in,
                              void* d_out, int out_size, void* d_ws, size_t ws_size,
                              hipStream_t stream) {
  const float* x     = (const float*)d_in[0];  // (4,2048,256) f32
  const float* coeff = (const float*)d_in[1];  // (8,13) f32
  const float* gw    = (const float*)d_in[2];  // (256,256,8) f32
  float* out = (float*)d_out;                  // (4,2048,256) f32
  _Float16* Gp = (_Float16*)d_ws;              // 256*2048 f16 = 1 MB

  kan_prep_gates<<<(NIN * NOUT) / 256, 256, 0, stream>>>(gw, Gp);
  kan_moe_fused<<<MTOT / MTILE, 256, 0, stream>>>(x, coeff, Gp, out);
}
